// SelfAttention_62775241998342
// MI455X (gfx1250) — matrix operand order repaired and checked by reading, not compiled, not run on hardware
//
#include <hip/hip_runtime.h>
#include <hip/hip_bf16.h>

// ---------------- problem constants (match reference) ----------------
#define Bc  8
#define Sc  1024
#define Ec  1024
#define Hc  8
#define DHc 128

// ---------------- vector types for WMMA ----------------
typedef _Float16 h4  __attribute__((ext_vector_type(4)));
typedef _Float16 h8  __attribute__((ext_vector_type(8)));
typedef _Float16 h16 __attribute__((ext_vector_type(16)));
typedef float    f4  __attribute__((ext_vector_type(4)));
typedef float    f8  __attribute__((ext_vector_type(8)));

union H16u { h16 v; h8 h[2]; };

__device__ inline f8 zero8() {
  f8 z = {0.f,0.f,0.f,0.f,0.f,0.f,0.f,0.f};
  return z;
}

// A-matrix fragment (16x32 f16). ISA layout: lanes 0-15 hold M=0..15 with
// K = {0..7, 16..23}; lanes 16-31 same M with K = {8..15, 24..31}.
__device__ inline h16 load_a_frag(const _Float16* rowp, int lane) {
  const int hb = (lane >> 4) * 8;
  H16u u;
  u.h[0] = *(const h8*)(rowp + hb);
  u.h[1] = *(const h8*)(rowp + hb + 16);
  return u.v;
}

// B-matrix fragment (32x16 f16). Lane holds column n = lane&15,
// K = {0..7, 16..23} for lanes 0-15, {8..15, 24..31} for lanes 16-31.
__device__ inline h16 load_b_frag(const _Float16* colp, int lane) {
  const int kb = (lane >> 4) * 8;
  H16u u;
  u.h[0] = *(const h8*)(colp + kb);
  u.h[1] = *(const h8*)(colp + kb + 16);
  return u.v;
}

__device__ inline f8 wmma32(h16 a, h16 b, f8 c) {
  return __builtin_amdgcn_wmma_f32_16x16x32_f16(
      false, a, false, b, (short)0, c, false, false);
}

// ---- CDNA5 async global->LDS copy (ASYNCcnt tracked, bypasses VGPRs) ----
// Each lane copies 16 bytes: lane supplies 32-bit LDS byte offset (low 32
// bits of the generic shared pointer, per the flat LDS-aperture rule) and a
// 64-bit global address.
__device__ inline void async_cp16(const void* lds_dst, const void* gsrc) {
  unsigned           ldso = (unsigned)(unsigned long long)lds_dst;
  unsigned long long gaddr = (unsigned long long)gsrc;
  asm volatile("global_load_async_to_lds_b128 %0, %1, off"
               :: "v"(ldso), "v"(gaddr) : "memory");
}
__device__ inline void async_wait0() {
  asm volatile("s_wait_asynccnt 0" ::: "memory");
}

// =====================================================================
// Kernel 1: projection GEMM. Q row-major [bh][s][d]; V stored TRANSPOSED
// [bh][d][s] so the attention kernel can async-copy it straight to LDS.
// grid.x = B*H*(S/128), grid.y: 0 -> Q, 1 -> V. 256 threads.
// =====================================================================
__global__ void __launch_bounds__(256)
proj_kernel(const float* __restrict__ hid,
            const float* __restrict__ Wq, const float* __restrict__ bq,
            const float* __restrict__ Wv, const float* __restrict__ bv,
            _Float16* __restrict__ Qh, _Float16* __restrict__ Vt) {
  __shared__ __align__(16) _Float16 As[128 * 40];   // hid tile [row][k]
  __shared__ __align__(16) _Float16 Bs[128 * 40];   // W tile^T [n][k]

  const int tid  = threadIdx.x;
  const int lane = tid & 31;
  const int wave = tid >> 5;
  const int ntiles = Sc / 128;
  const int bh    = blockIdx.x / ntiles;
  const int stile = blockIdx.x % ntiles;
  const int b = bh / Hc, h = bh % Hc;
  const int s0 = stile * 128;
  const int which = blockIdx.y;

  const float* W    = (which == 0 ? Wq : Wv) + (size_t)h * Ec * DHc;
  const float* bias = (which == 0 ? bq : bv) + h * DHc;

  f8 acc[8];
#pragma unroll
  for (int i = 0; i < 8; ++i) acc[i] = zero8();

  for (int k0 = 0; k0 < Ec; k0 += 32) {
    // ---- stage A: hid[s0..s0+127, k0..k0+31] f32 -> f16 ----
#pragma unroll
    for (int i = 0; i < 4; ++i) {
      int lin = tid + 256 * i;
      int r = lin >> 3, c4 = lin & 7;
      f4 v = *(const f4*)(hid + ((size_t)b * Sc + s0 + r) * Ec + k0 + c4 * 4);
      h4 hv = { (_Float16)v[0], (_Float16)v[1], (_Float16)v[2], (_Float16)v[3] };
      *(h4*)&As[r * 40 + c4 * 4] = hv;
    }
    // ---- stage B transposed: W[k0+kk, n] -> Bs[n][kk] ----
#pragma unroll
    for (int i = 0; i < 4; ++i) {
      int lin = tid + 256 * i;
      int kk = lin >> 5, n4 = lin & 31;
      f4 v = *(const f4*)(W + (size_t)(k0 + kk) * DHc + n4 * 4);
#pragma unroll
      for (int j = 0; j < 4; ++j) Bs[(n4 * 4 + j) * 40 + kk] = (_Float16)v[j];
    }
    if (k0 + 32 < Ec)
      __builtin_prefetch(hid + ((size_t)b * Sc + s0 + (tid >> 1)) * Ec + k0 + 32, 0, 1);
    __syncthreads();

    const int wr = wave * 16;
    h16 a = load_a_frag(&As[(wr + (lane & 15)) * 40], lane);
    h16 bf[8];
#pragma unroll
    for (int nt = 0; nt < 8; ++nt)
      bf[nt] = load_b_frag(&Bs[(nt * 16 + (lane & 15)) * 40], lane);
#pragma unroll
    for (int nt = 0; nt < 8; ++nt)
      acc[nt] = wmma32(a, bf[nt], acc[nt]);
    __syncthreads();
  }

  // ---- epilogue ----
  const int wr    = wave * 16;
  const int rbase = wr + (lane >> 4) * 8;           // 8 consecutive s rows
  if (which == 0) {
#pragma unroll
    for (int nt = 0; nt < 8; ++nt) {
      const int   n  = nt * 16 + (lane & 15);
      const float bz = bias[n];
#pragma unroll
      for (int j = 0; j < 8; ++j)
        Qh[((size_t)bh * Sc + s0 + rbase + j) * DHc + n] = (_Float16)(acc[nt][j] + bz);
    }
  } else {
    // transposed store: 8 consecutive s values at fixed d -> one b128 store
#pragma unroll
    for (int nt = 0; nt < 8; ++nt) {
      const int   n  = nt * 16 + (lane & 15);
      const float bz = bias[n];
      h8 hv;
#pragma unroll
      for (int j = 0; j < 8; ++j) hv[j] = (_Float16)(acc[nt][j] + bz);
      *(h8*)(Vt + ((size_t)bh * DHc + n) * Sc + s0 + rbase) = hv;
    }
  }
}

// =====================================================================
// Kernel 2: column-softmax stats via symmetry of Q Q^T.
// grid.x = B*H*(S/128), 256 threads.
// =====================================================================
__global__ void __launch_bounds__(256)
stats_kernel(const _Float16* __restrict__ Qh,
             float* __restrict__ Mb, float* __restrict__ Lb) {
  __shared__ __align__(16) _Float16 Qs[32 * 136];

  const int tid = threadIdx.x, lane = tid & 31, wave = tid >> 5;
  const int ntiles = Sc / 128;
  const int bh = blockIdx.x / ntiles;
  const int t0 = (blockIdx.x % ntiles) * 128;
  const int wr = wave * 16;
  const _Float16* Qbase = Qh + (size_t)bh * Sc * DHc;

  h16 aq[4];
  {
    const _Float16* rowp = Qbase + (size_t)(t0 + wr + (lane & 15)) * DHc;
#pragma unroll
    for (int kc = 0; kc < 4; ++kc) aq[kc] = load_a_frag(rowp + kc * 32, lane);
  }

  float m[8], l[8];
#pragma unroll
  for (int j = 0; j < 8; ++j) { m[j] = -3.0e38f; l[j] = 0.f; }

  for (int s0 = 0; s0 < Sc; s0 += 32) {
    // async-stage 32 key rows of Q (pure f16 copy, 16B per lane per op)
#pragma unroll
    for (int i = 0; i < 2; ++i) {
      int lin = tid + 256 * i;
      int sl = lin >> 4, c8 = lin & 15;
      async_cp16(&Qs[sl * 136 + c8 * 8],
                 Qbase + (size_t)(s0 + sl) * DHc + c8 * 8);
    }
    async_wait0();
    __syncthreads();

#pragma unroll
    for (int st = 0; st < 2; ++st) {
      const _Float16* colp = &Qs[(st * 16 + (lane & 15)) * 136];
      h16 bf[4];
#pragma unroll
      for (int kc = 0; kc < 4; ++kc) bf[kc] = load_b_frag(colp + kc * 32, lane);
      f8 c = zero8();
#pragma unroll
      for (int kc = 0; kc < 4; ++kc) c = wmma32(aq[kc], bf[kc], c);

#pragma unroll
      for (int j = 0; j < 8; ++j) {
        float v = c[j];
        float rmax = v;
        rmax = fmaxf(rmax, __shfl_xor(rmax, 1, 16));
        rmax = fmaxf(rmax, __shfl_xor(rmax, 2, 16));
        rmax = fmaxf(rmax, __shfl_xor(rmax, 4, 16));
        rmax = fmaxf(rmax, __shfl_xor(rmax, 8, 16));
        float nm = fmaxf(m[j], rmax);
        float e  = __expf(v - nm);
        e += __shfl_xor(e, 1, 16);
        e += __shfl_xor(e, 2, 16);
        e += __shfl_xor(e, 4, 16);
        e += __shfl_xor(e, 8, 16);
        l[j] = l[j] * __expf(m[j] - nm) + e;
        m[j] = nm;
      }
    }
    __syncthreads();
  }

  if ((lane & 15) == 0) {
    const int rbase = t0 + wr + (lane >> 4) * 8;
#pragma unroll
    for (int j = 0; j < 8; ++j) {
      Mb[(size_t)bh * Sc + rbase + j] = m[j];
      Lb[(size_t)bh * Sc + rbase + j] = l[j];
    }
  }
}

// =====================================================================
// Kernel 3: out[s,:] = relu( sum_t exp(q_s.q_t - M[t]) / L[t] * v[t,:] )
// grid.x = B*H*(S/128), 256 threads.
// =====================================================================
__global__ void __launch_bounds__(256)
attn_out_kernel(const _Float16* __restrict__ Qh, const _Float16* __restrict__ Vt,
                const float* __restrict__ Mb, const float* __restrict__ Lb,
                float* __restrict__ out) {
  __shared__ __align__(16) _Float16 Qt[32 * 136];     // key rows t0..t0+31
  __shared__ __align__(16) _Float16 Vs[128 * 40];     // V^T tile [d][t]
  __shared__ __align__(16) _Float16 Pl[8 * 16 * 40];  // per-wave P tile

  const int tid = threadIdx.x, lane = tid & 31, wave = tid >> 5;
  const int ntiles = Sc / 128;
  const int bh = blockIdx.x / ntiles;
  const int b = bh / Hc, h = bh % Hc;
  const int s0 = (blockIdx.x % ntiles) * 128;
  const int wr = wave * 16;
  const _Float16* Qbase = Qh + (size_t)bh * Sc * DHc;
  const _Float16* Vbase = Vt + (size_t)bh * DHc * Sc;   // transposed [d][s]

  h16 aq[4];
  {
    const _Float16* rowp = Qbase + (size_t)(s0 + wr + (lane & 15)) * DHc;
#pragma unroll
    for (int kc = 0; kc < 4; ++kc) aq[kc] = load_a_frag(rowp + kc * 32, lane);
  }

  f8 o[8];
#pragma unroll
  for (int i = 0; i < 8; ++i) o[i] = zero8();

  _Float16* Pw = &Pl[wave * 16 * 40];

  for (int t0 = 0; t0 < Sc; t0 += 32) {
    // ---- async-stage key rows Q[t0..t0+31] ----
#pragma unroll
    for (int i = 0; i < 2; ++i) {
      int lin = tid + 256 * i;
      int tl = lin >> 4, c8 = lin & 15;
      async_cp16(&Qt[tl * 136 + c8 * 8],
                 Qbase + (size_t)(t0 + tl) * DHc + c8 * 8);
    }
    // ---- async-stage V^T[d][t0..t0+31] (pure copy thanks to layout) ----
#pragma unroll
    for (int i = 0; i < 2; ++i) {
      int lin = tid + 256 * i;
      int d = lin >> 2, c8 = lin & 3;
      async_cp16(&Vs[d * 40 + c8 * 8],
                 Vbase + (size_t)d * Sc + t0 + c8 * 8);
    }
    // per-lane column stats: column t = t0 + st*16 + (lane&15)
    const float Ma = Mb[(size_t)bh * Sc + t0 + (lane & 15)];
    const float Ra = 1.0f / Lb[(size_t)bh * Sc + t0 + (lane & 15)];
    const float Mc = Mb[(size_t)bh * Sc + t0 + 16 + (lane & 15)];
    const float Rc = 1.0f / Lb[(size_t)bh * Sc + t0 + 16 + (lane & 15)];
    async_wait0();
    __syncthreads();

    // ---- scores -> P (f16, A-layout staging via LDS) ----
#pragma unroll
    for (int st = 0; st < 2; ++st) {
      const _Float16* colp = &Qt[(st * 16 + (lane & 15)) * 136];
      h16 bf[4];
#pragma unroll
      for (int kc = 0; kc < 4; ++kc) bf[kc] = load_b_frag(colp + kc * 32, lane);
      f8 c = zero8();
#pragma unroll
      for (int kc = 0; kc < 4; ++kc) c = wmma32(aq[kc], bf[kc], c);

      const float mm = st ? Mc : Ma;
      const float rr = st ? Rc : Ra;
      const int prow = (lane >> 4) * 8;
#pragma unroll
      for (int j = 0; j < 8; ++j) {
        float p = __expf(c[j] - mm) * rr;
        Pw[(prow + j) * 40 + st * 16 + (lane & 15)] = (_Float16)p;
      }
    }
    // wave-private LDS round trip: drain ds stores before re-reading
    asm volatile("s_wait_dscnt 0" ::: "memory");

    // ---- out += P(16x32) @ V(32x128) ----
    h16 ap = load_a_frag(&Pw[(lane & 15) * 40], lane);
    h16 bf[8];
#pragma unroll
    for (int nt = 0; nt < 8; ++nt)
      bf[nt] = load_b_frag(&Vs[(nt * 16 + (lane & 15)) * 40], lane);
#pragma unroll
    for (int nt = 0; nt < 8; ++nt)
      o[nt] = wmma32(ap, bf[nt], o[nt]);
    __syncthreads();
  }

  // ---- epilogue: ReLU, scatter heads back to [b, s, h*DH + d] ----
  const int rbase = wr + (lane >> 4) * 8;
#pragma unroll
  for (int nt = 0; nt < 8; ++nt) {
    const int col = h * DHc + nt * 16 + (lane & 15);
#pragma unroll
    for (int j = 0; j < 8; ++j) {
      const int s = s0 + rbase + j;
      out[((size_t)b * Sc + s) * Ec + col] = fmaxf(o[nt][j], 0.f);
    }
  }
}

// =====================================================================
extern "C" void kernel_launch(void* const* d_in, const int* in_sizes, int n_in,
                              void* d_out, int out_size, void* d_ws, size_t ws_size,
                              hipStream_t stream) {
  (void)in_sizes; (void)n_in; (void)out_size; (void)ws_size;
  const float* hid = (const float*)d_in[0];
  const float* Wq  = (const float*)d_in[1];
  const float* bq  = (const float*)d_in[2];
  const float* Wv  = (const float*)d_in[3];
  const float* bv  = (const float*)d_in[4];
  float* outp = (float*)d_out;

  // workspace: Qh (16MB f16, [bh][s][d]) | Vt (16MB f16, [bh][d][s]) | M | L
  const size_t qv_elems = (size_t)Bc * Hc * Sc * DHc;
  _Float16* Qh = (_Float16*)d_ws;
  _Float16* Vt = Qh + qv_elems;
  float*    Mb = (float*)(Vt + qv_elems);
  float*    Lb = Mb + (size_t)Bc * Hc * Sc;

  const int nblk = Bc * Hc * (Sc / 128);   // 512

  dim3 g1(nblk, 2, 1);
  proj_kernel<<<g1, 256, 0, stream>>>(hid, Wq, bq, Wv, bv, Qh, Vt);
  stats_kernel<<<nblk, 256, 0, stream>>>(Qh, Mb, Lb);
  attn_out_kernel<<<nblk, 256, 0, stream>>>(Qh, Vt, Mb, Lb, outp);
}
